// TargetEmbedding_86328842650205
// MI455X (gfx1250) — compile-verified, matrix operand order checked by reading, not run on hardware
//
#include <hip/hip_runtime.h>

// ---------------------------------------------------------------------------
// Problem constants (from the reference)
// ---------------------------------------------------------------------------
#define BB      32
#define NBB     16
#define CC      1024
#define HH      22
#define WW      22
#define PP      3
#define NROI    (BB * NBB)          // 512
#define KCONV   (CC * PP * PP)      // 9216
#define KSPLIT  4                   // split-K factor for the conv GEMM
#define KCHUNK  (KCONV / KSPLIT)    // 2304 (multiple of 32)
#define SCALE_F (1.0f / 16.0f)
#define EPS_F   1e-5f

typedef unsigned int u32;
typedef __attribute__((ext_vector_type(4)))  u32    u32x4;
typedef __attribute__((ext_vector_type(16))) __bf16 v16bf;
typedef __attribute__((ext_vector_type(8)))  float  v8f;

union Frag { u32x4 q[2]; v16bf v; };

__device__ __forceinline__ unsigned short f2bf(float f) {
    u32 u = __float_as_uint(f);
    u32 r = u + 0x7FFFu + ((u >> 16) & 1u);   // round-to-nearest-even
    return (unsigned short)(r >> 16);
}

__device__ __forceinline__ float hat_int(float t) {
    t = fminf(1.0f, fmaxf(-1.0f, t));
    return t + 0.5f - 0.5f * t * fabsf(t);
}

// ---------------------------------------------------------------------------
// K1: per-ROI separable integral weights Wx[512][3][22], Wy[512][3][22], 1/area
// ---------------------------------------------------------------------------
__global__ void roi_weights_kernel(const float* __restrict__ bbox,
                                   float* __restrict__ Wx,
                                   float* __restrict__ Wy,
                                   float* __restrict__ inv_area) {
    const int n = blockIdx.x;              // 0..511
    const int t = threadIdx.x;
    const float* bb = bbox + n * 4;
    const float x1 = bb[0] * SCALE_F;
    const float y1 = bb[1] * SCALE_F;
    const float x2 = (bb[0] + bb[2]) * SCALE_F;
    const float y2 = (bb[1] + bb[3]) * SCALE_F;
    const float bw = (x2 - x1) / (float)PP;
    const float bh = (y2 - y1) / (float)PP;
    if (t < PP * WW) {
        const int p = t / WW;
        const float fi = (float)(t % WW);
        const float xs = x1 + p * bw, xe = xs + bw;
        Wx[n * (PP * WW) + t] = hat_int(xe - fi) - hat_int(xs - fi);
        const float ys = y1 + p * bh, ye = ys + bh;
        Wy[n * (PP * HH) + t] = hat_int(ye - fi) - hat_int(ys - fi);
    }
    if (t == 0) {
        const float area = fmaxf(bw * bh, 0.0f);
        inv_area[n] = (area > 0.0f) ? (1.0f / fmaxf(area, 1e-12f)) : 0.0f;
    }
}

// ---------------------------------------------------------------------------
// K2: PrRoI pooling (fp32) + im2col pack to bf16 A matrix [512, 9216]
//     A[n][ci*9 + q*3 + p] = pooled / area   (q<->kh, p<->kw)
// ---------------------------------------------------------------------------
__global__ void pool_pack_kernel(const float* __restrict__ feat,
                                 const float* __restrict__ Wx,
                                 const float* __restrict__ Wy,
                                 const float* __restrict__ inv_area,
                                 unsigned short* __restrict__ Aout) {
    __shared__ float sWx[PP][WW];
    __shared__ float sWy[PP][HH];
    __shared__ float sIa;
    const int n = blockIdx.x;              // ROI index 0..511
    const int t = threadIdx.x;             // 256 threads
    if (t < PP * WW) {
        sWx[t / WW][t % WW] = Wx[n * (PP * WW) + t];
    } else if (t < 2 * PP * WW) {
        const int u = t - PP * WW;
        sWy[u / HH][u % HH] = Wy[n * (PP * HH) + u];
    }
    if (t == 2 * PP * WW) sIa = inv_area[n];
    __syncthreads();

    const int b = n >> 4;                  // n / NBB
    for (int ci = t; ci < CC; ci += 256) {
        const float* fp = feat + ((size_t)b * CC + ci) * (HH * WW);
        float out[PP][PP] = {};
        for (int h = 0; h < HH; ++h) {
            float r0 = 0.f, r1 = 0.f, r2 = 0.f;
            const float* fr = fp + h * WW;
            #pragma unroll
            for (int w = 0; w < WW; ++w) {
                const float f = fr[w];
                r0 += f * sWx[0][w];
                r1 += f * sWx[1][w];
                r2 += f * sWx[2][w];
            }
            const float y0 = sWy[0][h], y1 = sWy[1][h], y2 = sWy[2][h];
            out[0][0] += y0 * r0; out[0][1] += y0 * r1; out[0][2] += y0 * r2;
            out[1][0] += y1 * r0; out[1][1] += y1 * r1; out[1][2] += y1 * r2;
            out[2][0] += y2 * r0; out[2][1] += y2 * r1; out[2][2] += y2 * r2;
        }
        const float ia = sIa;
        unsigned short* ao = Aout + (size_t)n * KCONV + ci * 9;
        #pragma unroll
        for (int q = 0; q < PP; ++q)
            #pragma unroll
            for (int p = 0; p < PP; ++p)
                ao[q * PP + p] = f2bf(out[q][p] * ia);
    }
}

// ---------------------------------------------------------------------------
// K3: fp32 -> bf16 pack (weights)
// ---------------------------------------------------------------------------
__global__ void cvt_f32_bf16_kernel(const float* __restrict__ in,
                                    unsigned short* __restrict__ out, int count) {
    int i = blockIdx.x * blockDim.x + threadIdx.x;
    const int stride = gridDim.x * blockDim.x;
    for (; i < count; i += stride) out[i] = f2bf(in[i]);
}

// ---------------------------------------------------------------------------
// K4: fold conv bias + BN(eval) into per-channel scale/shift
// ---------------------------------------------------------------------------
__global__ void bn_fold_kernel(const float* __restrict__ cb, const float* __restrict__ g,
                               const float* __restrict__ be, const float* __restrict__ mn,
                               const float* __restrict__ vr,
                               float* __restrict__ scale, float* __restrict__ shift) {
    const int i = blockIdx.x * blockDim.x + threadIdx.x;
    if (i < CC) {
        const float s = g[i] * rsqrtf(vr[i] + EPS_F);
        scale[i] = s;
        shift[i] = (cb[i] - mn[i]) * s + be[i];
    }
}

// ---------------------------------------------------------------------------
// K5: bf16 WMMA GEMM   D[M,N] = A[M,K] x Bw[N,K]^T, fp32 accumulate.
//   Block = 128 threads (4 waves along N). Wave tile = 64(M) x 32(N),
//   block tile = 64 x 128. Requires M%64==0, N%128==0, kchunk%32==0.
//   gridDim.z == 1 : fused epilogue  y = acc*scale[col]+shift[col] (+ReLU),
//                    writes bf16 activation and/or fp32 output.
//   gridDim.z  > 1 : split-K, raw fp32 partials -> partial[z][M][N].
//
//   Single fragment set per 32-wide K slab; the scheduler clause-groups the
//   b128 loads and interleaves them with the WMMAs (round-1 codegen was
//   spill-free and self-pipelined; manual double-buffering caused scratch
//   spills and is deliberately NOT used).
//
//   Fragment staging per CDNA5 ISA tables (16-bit A 16x32, D f32 16x16):
//     A: lane l -> row (l&15); K chunks {base+(l>>4)*8 .. +7} and {+16 .. +23}
//        as two contiguous 16B loads. B symmetric from [N,K] row-major.
//     D: VGPR v -> row v + 8*(l>>4), col (l&15).
// ---------------------------------------------------------------------------
__global__ void gemm_bf16_kernel(const unsigned short* __restrict__ A,
                                 const unsigned short* __restrict__ Bw,
                                 const float* __restrict__ scale,
                                 const float* __restrict__ shift,
                                 unsigned short* __restrict__ outB,
                                 float* __restrict__ outF,
                                 float* __restrict__ partial,
                                 int M, int N, int K, int kchunk, int relu) {
    const int lane  = threadIdx.x & 31;
    const int wave  = threadIdx.x >> 5;
    const int lmod  = lane & 15;
    const int lhalf = lane >> 4;
    const int m0 = blockIdx.y * 64;
    const int n0 = blockIdx.x * 128 + wave * 32;
    const int k0   = blockIdx.z * kchunk;
    const int kend = (k0 + kchunk < K) ? (k0 + kchunk) : K;

    const unsigned short* Ap[4];
    #pragma unroll
    for (int f = 0; f < 4; ++f)
        Ap[f] = A + (size_t)(m0 + f * 16 + lmod) * K + lhalf * 8;
    const unsigned short* Bp[2];
    #pragma unroll
    for (int g = 0; g < 2; ++g)
        Bp[g] = Bw + (size_t)(n0 + g * 16 + lmod) * K + lhalf * 8;

    v8f acc[4][2] = {};

    for (int kb = k0; kb < kend; kb += 32) {
        Frag a[4], b[2];
        #pragma unroll
        for (int f = 0; f < 4; ++f) {
            a[f].q[0] = *(const u32x4*)(Ap[f] + kb);
            a[f].q[1] = *(const u32x4*)(Ap[f] + kb + 16);
        }
        #pragma unroll
        for (int g = 0; g < 2; ++g) {
            b[g].q[0] = *(const u32x4*)(Bp[g] + kb);
            b[g].q[1] = *(const u32x4*)(Bp[g] + kb + 16);
        }
        if (kb + 32 < kend) {   // prefetch next K-slab (global_prefetch_b8)
            #pragma unroll
            for (int f = 0; f < 4; ++f) __builtin_prefetch(Ap[f] + kb + 32, 0, 3);
            #pragma unroll
            for (int g = 0; g < 2; ++g) __builtin_prefetch(Bp[g] + kb + 32, 0, 3);
        }
        #pragma unroll
        for (int f = 0; f < 4; ++f)
            #pragma unroll
            for (int g = 0; g < 2; ++g)
                acc[f][g] = __builtin_amdgcn_wmma_f32_16x16x32_bf16(
                    false, a[f].v, false, b[g].v, (short)0, acc[f][g],
                    false, false);
    }

    if (gridDim.z > 1) {
        // split-K: raw fp32 partials
        float* pp = partial + (size_t)blockIdx.z * M * N;
        #pragma unroll
        for (int f = 0; f < 4; ++f)
            #pragma unroll
            for (int g = 0; g < 2; ++g) {
                const int col = n0 + g * 16 + lmod;
                #pragma unroll
                for (int v = 0; v < 8; ++v) {
                    const int row = m0 + f * 16 + lhalf * 8 + v;
                    pp[(size_t)row * N + col] = acc[f][g][v];
                }
            }
    } else {
        // fused epilogue
        #pragma unroll
        for (int f = 0; f < 4; ++f)
            #pragma unroll
            for (int g = 0; g < 2; ++g) {
                const int col = n0 + g * 16 + lmod;
                const float sc = scale ? scale[col] : 1.0f;
                const float sh = shift[col];
                #pragma unroll
                for (int v = 0; v < 8; ++v) {
                    const int row = m0 + f * 16 + lhalf * 8 + v;
                    float x = acc[f][g][v] * sc + sh;
                    if (relu) x = fmaxf(x, 0.0f);
                    if (outF) outF[(size_t)row * N + col] = x;
                    if (outB) outB[(size_t)row * N + col] = f2bf(x);
                }
            }
    }
}

// ---------------------------------------------------------------------------
// K6: split-K reduction + scale/shift/ReLU + bf16 pack
// ---------------------------------------------------------------------------
__global__ void reduce_epilogue_kernel(const float* __restrict__ partial,
                                       int S, int M, int N,
                                       const float* __restrict__ scale,
                                       const float* __restrict__ shift,
                                       int relu,
                                       unsigned short* __restrict__ outB,
                                       float* __restrict__ outF) {
    const size_t total = (size_t)M * N;
    size_t i = (size_t)blockIdx.x * blockDim.x + threadIdx.x;
    const size_t stride = (size_t)gridDim.x * blockDim.x;
    for (; i < total; i += stride) {
        float s = 0.0f;
        for (int t = 0; t < S; ++t) s += partial[(size_t)t * total + i];
        const int col = (int)(i % (size_t)N);
        float x = s * (scale ? scale[col] : 1.0f) + shift[col];
        if (relu) x = fmaxf(x, 0.0f);
        if (outF) outF[i] = x;
        if (outB) outB[i] = f2bf(x);
    }
}

// ---------------------------------------------------------------------------
// Host launcher
// ---------------------------------------------------------------------------
extern "C" void kernel_launch(void* const* d_in, const int* in_sizes, int n_in,
                              void* d_out, int out_size, void* d_ws, size_t ws_size,
                              hipStream_t stream) {
    (void)in_sizes; (void)n_in; (void)out_size; (void)ws_size;
    const float* feat   = (const float*)d_in[0];
    const float* bbox   = (const float*)d_in[1];
    const float* conv_w = (const float*)d_in[2];
    const float* conv_b = (const float*)d_in[3];
    const float* gamma  = (const float*)d_in[4];
    const float* beta   = (const float*)d_in[5];
    const float* mean   = (const float*)d_in[6];
    const float* var    = (const float*)d_in[7];
    const float* w1     = (const float*)d_in[8];
    const float* b1     = (const float*)d_in[9];
    const float* w2     = (const float*)d_in[10];
    const float* b2     = (const float*)d_in[11];
    const float* w3     = (const float*)d_in[12];
    const float* b3     = (const float*)d_in[13];

    char* ws = (char*)d_ws;
    size_t off = 0;
    auto carve = [&](size_t bytes) -> char* {
        off = (off + 255) & ~(size_t)255;
        char* p = ws + off;
        off += bytes;
        return p;
    };

    float*          Wx    = (float*)carve((size_t)NROI * PP * WW * 4);
    float*          Wy    = (float*)carve((size_t)NROI * PP * HH * 4);
    float*          ia    = (float*)carve((size_t)NROI * 4);
    float*          scl   = (float*)carve((size_t)CC * 4);
    float*          shf   = (float*)carve((size_t)CC * 4);
    unsigned short* Aconv = (unsigned short*)carve((size_t)NROI * KCONV * 2);
    unsigned short* Wc    = (unsigned short*)carve((size_t)CC * KCONV * 2);
    unsigned short* X2    = (unsigned short*)carve((size_t)NROI * CC * 2);
    unsigned short* W1b   = (unsigned short*)carve((size_t)512 * CC * 2);
    unsigned short* X3    = (unsigned short*)carve((size_t)NROI * 512 * 2);
    unsigned short* W2b   = (unsigned short*)carve((size_t)512 * 512 * 2);
    unsigned short* X4    = (unsigned short*)carve((size_t)NROI * 512 * 2);
    unsigned short* W3b   = (unsigned short*)carve((size_t)512 * 512 * 2);
    float*          Pbuf  = (float*)carve((size_t)KSPLIT * NROI * CC * 4);

    // 1) ROI integral weights
    roi_weights_kernel<<<NROI, 96, 0, stream>>>(bbox, Wx, Wy, ia);

    // 2) pooling + bf16 im2col pack
    pool_pack_kernel<<<NROI, 256, 0, stream>>>(feat, Wx, Wy, ia, Aconv);

    // 3) weight packs to bf16
    cvt_f32_bf16_kernel<<<2048, 256, 0, stream>>>(conv_w, Wc, CC * KCONV);
    cvt_f32_bf16_kernel<<<512, 256, 0, stream>>>(w1, W1b, 512 * CC);
    cvt_f32_bf16_kernel<<<256, 256, 0, stream>>>(w2, W2b, 512 * 512);
    cvt_f32_bf16_kernel<<<256, 256, 0, stream>>>(w3, W3b, 512 * 512);

    // 4) fold conv bias + BN into scale/shift
    bn_fold_kernel<<<(CC + 255) / 256, 256, 0, stream>>>(conv_b, gamma, beta,
                                                         mean, var, scl, shf);

    // 5) GEMM chain (all v_wmma_f32_16x16x32_bf16, fp32 accumulate)
    // conv-as-GEMM: [512,9216] x [9216,1024], split-K=4 -> fp32 partials
    gemm_bf16_kernel<<<dim3(CC / 128, NROI / 64, KSPLIT), 128, 0, stream>>>(
        Aconv, Wc, nullptr, nullptr, nullptr, nullptr, Pbuf,
        NROI, CC, KCONV, KCHUNK, 0);
    // reduce partials + BN + ReLU -> X2 (bf16)
    reduce_epilogue_kernel<<<2048, 256, 0, stream>>>(
        Pbuf, KSPLIT, NROI, CC, scl, shf, 1, X2, nullptr);
    // fc1: [512,1024] x [1024,512] -> +b1, ReLU -> X3
    gemm_bf16_kernel<<<dim3(512 / 128, NROI / 64, 1), 128, 0, stream>>>(
        X2, W1b, nullptr, b1, X3, nullptr, nullptr, NROI, 512, CC, CC, 1);
    // fc2: [512,512] x [512,512] -> +b2, ReLU -> X4
    gemm_bf16_kernel<<<dim3(512 / 128, NROI / 64, 1), 128, 0, stream>>>(
        X3, W2b, nullptr, b2, X4, nullptr, nullptr, NROI, 512, 512, 512, 1);
    // fc3: [512,512] x [512,512] -> +b3 -> d_out (fp32, == [B,NBB,512])
    gemm_bf16_kernel<<<dim3(512 / 128, NROI / 64, 1), 128, 0, stream>>>(
        X4, W3b, nullptr, b3, nullptr, (float*)d_out, nullptr,
        NROI, 512, 512, 512, 0);
}
